// LinearAttention_53326313947410
// MI455X (gfx1250) — matrix operand order repaired and checked by reading, not compiled, not run on hardware
//
#include <hip/hip_runtime.h>
#include <hip/hip_bf16.h>

typedef __attribute__((ext_vector_type(16))) _Float16 v16h;
typedef __attribute__((ext_vector_type(8)))  float    v8f;

#define HEADS    4
#define DHEAD    32
#define CIN      64
#define HIDDEN   128
#define NQKV     384
#define NMEM     4
#define NVOX     110592   // 48^3
#define BATCH    2
#define NTILE    16
#define TILES_PER_B (NVOX / NTILE)         // 6912
#define TILE_ELEMS  (HIDDEN * NTILE)       // 2048 f16 = 4KB per voxel tile
#define CTX_NBLK 108                       // NVOX / 1024
#define CTX_PART (DHEAD*DHEAD + DHEAD)     // 1056 floats: ctx[32][32] + Z[32]

// ---------------------------------------------------------------- helpers

__device__ __forceinline__ v8f wmma_f16(v16h a, v16h b, v8f c) {
  // D = A(16x32 f16) * B(32x16 f16) + C(16x16 f32)
  return __builtin_amdgcn_wmma_f32_16x16x32_f16(false, a, false, b,
                                                (short)0, c, false, false);
}

__device__ __forceinline__ v8f zero8() {
  v8f z;
#pragma unroll
  for (int i = 0; i < 8; ++i) z[i] = 0.f;
  return z;
}

// A-fragment K mapping (16-bit A 16x32): kb in {0,8}; halves 0-7 -> kb+0..7,
// halves 8-15 -> kb+16..23  (per ISA 7.12.2)
__device__ __forceinline__ int kmapA(int lane, int h) {
  return ((lane < 16) ? 0 : 8) + h + ((h >= 8) ? 8 : 0);
}
// B-fragment K mapping (16-bit B 32x16): same as A; kb in {0,8}; halves 0-7 -> kb+0..7,
// halves 8-15 -> kb+16..23
__device__ __forceinline__ int kmapB(int lane, int h) {
  return ((lane < 16) ? 0 : 8) + h + ((h >= 8) ? 8 : 0);
}

// load A fragment from row-major f16 matrix (leading dim ld), k offset k0
__device__ __forceinline__ v16h loadA(const _Float16* base, int row0, int ld,
                                      int k0, int lane) {
  v16h a;
  const _Float16* p = base + (size_t)(row0 + (lane & 15)) * ld + k0;
#pragma unroll
  for (int h = 0; h < 16; ++h) a[h] = p[kmapA(lane, h)];
  return a;
}

// ds_swizzle butterfly max over 16-lane halves (xor 1,2,4,8)
__device__ __forceinline__ float swz_max16(float v) {
  v = fmaxf(v, __int_as_float(__builtin_amdgcn_ds_swizzle(__float_as_int(v), 0x041F)));
  v = fmaxf(v, __int_as_float(__builtin_amdgcn_ds_swizzle(__float_as_int(v), 0x081F)));
  v = fmaxf(v, __int_as_float(__builtin_amdgcn_ds_swizzle(__float_as_int(v), 0x101F)));
  v = fmaxf(v, __int_as_float(__builtin_amdgcn_ds_swizzle(__float_as_int(v), 0x201F)));
  return v;
}

// exact (order-independent) float atomic max
__device__ __forceinline__ void atomicMaxF(float* addr, float val) {
  if (val >= 0.f) atomicMax((int*)addr, __float_as_int(val));
  else            atomicMin((unsigned int*)addr, __float_as_uint(val));
}

// ---------------------------------------------------------------- kernel 1

__global__ void k_init(const float* __restrict__ wqkv,
                       const float* __restrict__ wout,
                       _Float16* __restrict__ wqkv_h,
                       _Float16* __restrict__ wout_h,
                       float* __restrict__ rowmax) {
  int i = blockIdx.x * 256 + threadIdx.x;
  if (i < NQKV * CIN)     wqkv_h[i] = (_Float16)wqkv[i];
  if (i < CIN * HIDDEN)   wout_h[i] = (_Float16)wout[i];
  if (i < BATCH * HIDDEN) rowmax[i] = -3.4e38f;
}

// ------------------------------------------------- kernel 2: rmsnorm + QKV
// q/k/v stored tile-major: [tile][row(0..127)][col(0..15)], 4KB per tile.

__global__ __launch_bounds__(128)
void k_qkv(const float* __restrict__ x, const float* __restrict__ g_in,
           const _Float16* __restrict__ wqkv_h,
           _Float16* __restrict__ qh, _Float16* __restrict__ kh,
           _Float16* __restrict__ vh, float* __restrict__ rowmax) {
  __shared__ float    sx[CIN][NTILE + 1];
  __shared__ _Float16 xh[CIN][NTILE + 8];
  __shared__ float    cscale[NTILE];
  const int tid = threadIdx.x;
  const int tj = blockIdx.x;                 // global tile (includes batch)
  const int bb = tj / TILES_PER_B;
  const int j0 = (tj % TILES_PER_B) * NTILE;

  {
    const int col = tid & 15, r0 = tid >> 4;
#pragma unroll
    for (int i = 0; i < 8; ++i) {
      int r = r0 + 8 * i;
      sx[r][col] = x[(size_t)(bb * CIN + r) * NVOX + j0 + col];
    }
  }
  __syncthreads();
  if (tid < NTILE) {
    float s = 0.f;
    for (int r = 0; r < CIN; ++r) { float v = sx[r][tid]; s += v * v; }
    cscale[tid] = 8.0f / fmaxf(sqrtf(s), 1e-12f);   // sqrt(64)/||x||
  }
  __syncthreads();
  {
    const int col = tid & 15, r0 = tid >> 4;
#pragma unroll
    for (int i = 0; i < 8; ++i) {
      int r = r0 + 8 * i;
      xh[r][col] = (_Float16)(sx[r][col] * cscale[col] * g_in[r]);
    }
  }
  __syncthreads();

  const int wid = tid >> 5, lane = tid & 31;
  for (int t = 0; t < 6; ++t) {               // 4 waves x 6 = 24 row tiles
    const int rt = wid * 6 + t;
    v8f acc = zero8();
#pragma unroll
    for (int kk = 0; kk < CIN; kk += 32) {
      v16h a = loadA(wqkv_h, rt * 16, CIN, kk, lane);
      v16h b;
      const int nn = lane & 15;
#pragma unroll
      for (int h = 0; h < 16; ++h) b[h] = xh[kk + kmapB(lane, h)][nn];
      acc = wmma_f16(a, b, acc);
    }
    const int o0 = rt * 16;
    const int sect = o0 >> 7;                  // 0:q 1:k 2:v
    const int obase = o0 & 127;
    const int col = lane & 15;
    const int rhi = (lane & 16) ? 8 : 0;
    _Float16* dst = (sect == 0) ? qh : ((sect == 1) ? kh : vh);
    dst += (size_t)tj * TILE_ELEMS;
#pragma unroll
    for (int r = 0; r < 8; ++r) {
      int row = obase + r + rhi;
      dst[row * NTILE + col] = (_Float16)acc[r];
    }
    if (sect == 1) {                           // sequence-softmax row maxima
#pragma unroll
      for (int r = 0; r < 8; ++r) {
        float v = swz_max16(acc[r]);
        if (col == 0)
          atomicMaxF(&rowmax[bb * HIDDEN + obase + r + rhi], v);
      }
    }
  }
}

// -------------------------------- kernel 3: fold mem_k columns into rowmax

__global__ void k_foldmem(const float* __restrict__ mem_kv,
                          float* __restrict__ rowmax) {
  int i = threadIdx.x;
  if (i < BATCH * HIDDEN) {
    int hd = i & 127, h = hd >> 5, d = hd & 31;
    float m = rowmax[i];
    for (int mm = 0; mm < NMEM; ++mm)
      m = fmaxf(m, mem_kv[((0 * HEADS + h) * DHEAD + d) * NMEM + mm]);
    rowmax[i] = m;
  }
}

// ----------------------- kernel 4: context = sum_n exp(k-max) * v^T  (WMMA)

__global__ __launch_bounds__(128)
void k_ctx(const _Float16* __restrict__ kh, const _Float16* __restrict__ vh,
           const float* __restrict__ rowmax, float* __restrict__ ctxpart) {
  __shared__ float part[4][CTX_PART];
  const int tid = threadIdx.x, wid = tid >> 5, lane = tid & 31;
  const int bh = blockIdx.y, bb = bh >> 2, h = bh & 3;
  const int rowlo = h * DHEAD;                // head's row slice within a tile

  v8f acc[2][2], accZ[2];
#pragma unroll
  for (int i = 0; i < 2; ++i) {
    accZ[i] = zero8();
#pragma unroll
    for (int j = 0; j < 2; ++j) acc[i][j] = zero8();
  }
  v16h ones;
#pragma unroll
  for (int i = 0; i < 16; ++i) ones[i] = (_Float16)1.0f;

  const int m = lane & 15;
  const float mx0 = rowmax[bb * HIDDEN + rowlo + m];
  const float mx1 = rowmax[bb * HIDDEN + rowlo + 16 + m];

  for (int c = 0; c < 8; ++c) {               // 8 chunks of 32 columns/wave
    const int n0 = blockIdx.x * 1024 + wid * 256 + c * 32;
    const size_t t0 = (size_t)(bb * TILES_PER_B + (n0 >> 4));  // 2 tiles/chunk
    // A fragments: exp(k - max); kmapA splits into two 8-half runs, one per tile
    v16h ea[2];
#pragma unroll
    for (int dt = 0; dt < 2; ++dt) {
      const int row = rowlo + dt * 16 + m;
      const int kb = (lane < 16) ? 0 : 8;
      const _Float16* kp0 = kh + (t0 + 0) * TILE_ELEMS + row * NTILE + kb;
      const _Float16* kp1 = kh + (t0 + 1) * TILE_ELEMS + row * NTILE + kb;
      const float mx = dt ? mx1 : mx0;
      if (c < 7) {                            // stream prefetch, next chunk
        __builtin_prefetch(kp0 + 2 * TILE_ELEMS, 0, 3);
        __builtin_prefetch(kp1 + 2 * TILE_ELEMS, 0, 3);
      }
#pragma unroll
      for (int hh = 0; hh < 8; ++hh)
        ea[dt][hh] = (_Float16)__expf((float)kp0[hh] - mx);
#pragma unroll
      for (int hh = 0; hh < 8; ++hh)
        ea[dt][8 + hh] = (_Float16)__expf((float)kp1[hh] - mx);
    }
    // B fragments: v rows, two 8-half runs per lane, one per tile (matches A)
    v16h vb[2];
#pragma unroll
    for (int et = 0; et < 2; ++et) {
      const int row = rowlo + et * 16 + m;
      const _Float16* vp = vh + t0 * TILE_ELEMS +
                           row * NTILE + ((lane < 16) ? 0 : 8);
      if (c < 7) __builtin_prefetch(vp + 2 * TILE_ELEMS, 0, 3);
#pragma unroll
      for (int hh = 0; hh < 16; ++hh)
        vb[et][hh] = vp[(hh < 8) ? hh : (TILE_ELEMS + hh - 8)];
    }
#pragma unroll
    for (int dt = 0; dt < 2; ++dt) {
#pragma unroll
      for (int et = 0; et < 2; ++et)
        acc[dt][et] = wmma_f16(ea[dt], vb[et], acc[dt][et]);
      accZ[dt] = wmma_f16(ea[dt], ones, accZ[dt]);   // Z = row sums
    }
  }

  {
    const int col = lane & 15, rhi = (lane & 16) ? 8 : 0;
#pragma unroll
    for (int dt = 0; dt < 2; ++dt) {
#pragma unroll
      for (int et = 0; et < 2; ++et)
#pragma unroll
        for (int r = 0; r < 8; ++r)
          part[wid][(dt * 16 + r + rhi) * DHEAD + et * 16 + col] =
              acc[dt][et][r];
      if (col == 0)
#pragma unroll
        for (int r = 0; r < 8; ++r)
          part[wid][DHEAD * DHEAD + dt * 16 + r + rhi] = accZ[dt][r];
    }
  }
  __syncthreads();
  float* dst = ctxpart + ((size_t)bh * CTX_NBLK + blockIdx.x) * CTX_PART;
  for (int idx = tid; idx < CTX_PART; idx += 128)
    dst[idx] = part[0][idx] + part[1][idx] + part[2][idx] + part[3][idx];
}

// -------------- kernel 5: reduce partials + mem-kv + normalize -> ctx^T f16

__global__ __launch_bounds__(256)
void k_ctxfin(const float* __restrict__ ctxpart,
              const float* __restrict__ mem_kv,
              const float* __restrict__ rowmax, _Float16* __restrict__ ctxT) {
  __shared__ float Z[DHEAD];
  const int bh = blockIdx.x, bb = bh >> 2, h = bh & 3;
  const int tid = threadIdx.x;
  const float* base = ctxpart + (size_t)bh * CTX_NBLK * CTX_PART;
  if (tid < DHEAD) {
    const float mx = rowmax[bb * HIDDEN + h * DHEAD + tid];
    float s = 0.f;
    for (int blk = 0; blk < CTX_NBLK; ++blk)
      s += base[blk * CTX_PART + DHEAD * DHEAD + tid];
    for (int mm = 0; mm < NMEM; ++mm)
      s += __expf(mem_kv[((0 * HEADS + h) * DHEAD + tid) * NMEM + mm] - mx);
    Z[tid] = s;
  }
  __syncthreads();
  for (int idx = tid; idx < DHEAD * DHEAD; idx += 256) {
    const int d = idx >> 5, e = idx & 31;
    float s = 0.f;
    for (int blk = 0; blk < CTX_NBLK; ++blk) s += base[blk * CTX_PART + idx];
    const float mx = rowmax[bb * HIDDEN + h * DHEAD + d];
    for (int mm = 0; mm < NMEM; ++mm)
      s += __expf(mem_kv[((0 * HEADS + h) * DHEAD + d) * NMEM + mm] - mx) *
           mem_kv[((1 * HEADS + h) * DHEAD + e) * NMEM + mm];
    ctxT[(size_t)bh * DHEAD * DHEAD + e * DHEAD + d] = (_Float16)(s / Z[d]);
  }
}

// --------- kernel 6: softmax_d(q), ctx^T @ q, w_out GEMM, bias, rmsnorm

__global__ __launch_bounds__(128)
void k_out(const _Float16* __restrict__ qh, const _Float16* __restrict__ ctxT,
           const _Float16* __restrict__ wout_h,
           const float* __restrict__ b_out, const float* __restrict__ g_out,
           float* __restrict__ out) {
  __shared__ _Float16 qraw[TILE_ELEMS];            // raw q tile, contiguous
  __shared__ _Float16 qsh[HIDDEN][NTILE + 2];      // softmaxed, padded for B
  __shared__ _Float16 hidsh[HIDDEN][NTILE + 2];
  __shared__ float colred[4][NTILE];
  const int tid = threadIdx.x, wid = tid >> 5, lane = tid & 31;
  const int tj = blockIdx.x;
  const int bb = tj / TILES_PER_B;
  const int j0 = (tj % TILES_PER_B) * NTILE;

  {                                           // bulk-load 4KB q tile
    const _Float16* qsrc = qh + (size_t)tj * TILE_ELEMS;
#pragma unroll
    for (int i = 0; i < 16; ++i) qraw[tid * 16 + i] = qsrc[tid * 16 + i];
  }
  __syncthreads();

  if (tid < 64) {                             // softmax over d per (head,col)
    const int h = tid >> 4, col = tid & 15;
    float vals[DHEAD];
    float mx = -3.4e38f;
    for (int d = 0; d < DHEAD; ++d) {
      vals[d] = (float)qraw[(h * DHEAD + d) * NTILE + col];
      mx = fmaxf(mx, vals[d]);
    }
    float s = 0.f;
    for (int d = 0; d < DHEAD; ++d) { vals[d] = __expf(vals[d] - mx); s += vals[d]; }
    const float sc = 0.1767766952966369f / s;   // DIM_HEAD^-0.5 / sum
    for (int d = 0; d < DHEAD; ++d)
      qsh[h * DHEAD + d][col] = (_Float16)(vals[d] * sc);
  }
  __syncthreads();

#pragma unroll
  for (int t = 0; t < 2; ++t) {               // hidden = ctx^T @ q_soft
    const int tile = wid * 2 + t;             // 8 tiles: (head, e-half)
    const int h = tile >> 1, et = tile & 1;
    const int bh = bb * HEADS + h;
    v16h a = loadA(ctxT + (size_t)bh * DHEAD * DHEAD, et * 16, DHEAD, 0, lane);
    v16h bfr;
    const int nn = lane & 15;
#pragma unroll
    for (int hh = 0; hh < 16; ++hh) bfr[hh] = qsh[h * DHEAD + kmapB(lane, hh)][nn];
    v8f acc = wmma_f16(a, bfr, zero8());
    const int col = lane & 15, rhi = (lane & 16) ? 8 : 0;
#pragma unroll
    for (int r = 0; r < 8; ++r)
      hidsh[h * DHEAD + et * 16 + r + rhi][col] = (_Float16)acc[r];
  }
  __syncthreads();

  v8f acc = zero8();                          // out64 = w_out @ hidden
#pragma unroll
  for (int kk = 0; kk < HIDDEN; kk += 32) {
    v16h a = loadA(wout_h, wid * 16, HIDDEN, kk, lane);
    v16h bfr;
    const int nn = lane & 15;
#pragma unroll
    for (int hh = 0; hh < 16; ++hh) bfr[hh] = hidsh[kk + kmapB(lane, hh)][nn];
    acc = wmma_f16(a, bfr, acc);
  }
  const int col = lane & 15, rhi = (lane & 16) ? 8 : 0;
  float ov[8];
  float ss = 0.f;
#pragma unroll
  for (int r = 0; r < 8; ++r) {
    const int c = wid * 16 + r + rhi;
    ov[r] = acc[r] + b_out[c];
    ss += ov[r] * ov[r];
  }
  // combine the two 16-lane halves (same column): ds_swizzle xor-16
  ss += __int_as_float(__builtin_amdgcn_ds_swizzle(__float_as_int(ss), 0x401F));
  if (lane < 16) colred[wid][lane] = ss;
  __syncthreads();
  const float tot = colred[0][col] + colred[1][col] + colred[2][col] + colred[3][col];
  const float inv = 8.0f / fmaxf(sqrtf(tot), 1e-12f);
#pragma unroll
  for (int r = 0; r < 8; ++r) {
    const int c = wid * 16 + r + rhi;
    out[(size_t)(bb * CIN + c) * NVOX + j0 + col] = ov[r] * inv * g_out[c];
  }
}

// ---------------------------------------------------------------- launcher

extern "C" void kernel_launch(void* const* d_in, const int* in_sizes, int n_in,
                              void* d_out, int out_size, void* d_ws,
                              size_t ws_size, hipStream_t stream) {
  const float* x      = (const float*)d_in[0];
  const float* g_in   = (const float*)d_in[1];
  const float* w_qkv  = (const float*)d_in[2];
  const float* mem_kv = (const float*)d_in[3];
  const float* w_out  = (const float*)d_in[4];
  const float* b_out  = (const float*)d_in[5];
  const float* g_out  = (const float*)d_in[6];
  float* out = (float*)d_out;
  (void)in_sizes; (void)n_in; (void)out_size; (void)ws_size;

  char* p = (char*)d_ws;
  size_t off = 0;
  auto take = [&](size_t bytes) -> void* {
    void* r = p + off;
    off += (bytes + 255) & ~(size_t)255;
    return r;
  };
  const size_t qkvElems = (size_t)BATCH * HIDDEN * NVOX;
  _Float16* qh      = (_Float16*)take(qkvElems * 2);
  _Float16* kh      = (_Float16*)take(qkvElems * 2);
  _Float16* vh      = (_Float16*)take(qkvElems * 2);
  _Float16* wqkv_h  = (_Float16*)take((size_t)NQKV * CIN * 2);
  _Float16* wout_h  = (_Float16*)take((size_t)CIN * HIDDEN * 2);
  float*    rowmax  = (float*)take((size_t)BATCH * HIDDEN * 4);
  float*    ctxpart = (float*)take((size_t)BATCH * HEADS * CTX_NBLK * CTX_PART * 4);
  _Float16* ctxT    = (_Float16*)take((size_t)BATCH * HEADS * DHEAD * DHEAD * 2);

  k_init<<<(NQKV * CIN + 255) / 256, 256, 0, stream>>>(w_qkv, w_out, wqkv_h,
                                                       wout_h, rowmax);
  k_qkv<<<BATCH * TILES_PER_B, 128, 0, stream>>>(x, g_in, wqkv_h, qh, kh, vh,
                                                 rowmax);
  k_foldmem<<<1, 256, 0, stream>>>(mem_kv, rowmax);
  dim3 gctx(CTX_NBLK, BATCH * HEADS);
  k_ctx<<<gctx, 128, 0, stream>>>(kh, vh, rowmax, ctxpart);
  k_ctxfin<<<BATCH * HEADS, 256, 0, stream>>>(ctxpart, mem_kv, rowmax, ctxT);
  k_out<<<BATCH * TILES_PER_B, 128, 0, stream>>>(qh, ctxT, wout_h, b_out,
                                                 g_out, out);
}
